// NodeClusterReconstruction_44444321579619
// MI455X (gfx1250) — matrix operand order repaired and checked by reading, not compiled, not run on hardware
//
#include <hip/hip_runtime.h>
#include <hip/hip_bf16.h>
#include <math.h>

// ---------------------------------------------------------------------------
// NodeClusterReconstruction for MI455X (gfx1250, wave32, WMMA + async-to-LDS).
// All matmuls run through v_wmma_f32_16x16x32_bf16 (f32 accumulate).
// Shared tiles staged with GLOBAL_LOAD_ASYNC_TO_LDS_B128 (ASYNCcnt).
// ---------------------------------------------------------------------------

typedef __attribute__((ext_vector_type(8)))  __bf16 v8bf;
typedef __attribute__((ext_vector_type(16))) __bf16 v16bf;
typedef __attribute__((ext_vector_type(8)))  float  v8f;

#define DIMD  256
#define NQ    4096
#define NKV   256
#define BATCH 8
#define NH    4
#define HD    64

__device__ __forceinline__ v8f wmma_bf16(v16bf a, v16bf b, v8f c) {
  // 8 args: (neg_a, A, neg_b, B, c_mod, C, reuse_a, reuse_b)
  return __builtin_amdgcn_wmma_f32_16x16x32_bf16(false, a, false, b, (short)0, c,
                                                 false, false);
}

// Low 32 bits of a generic pointer into LDS = workgroup-relative LDS byte
// address (ISA 10.2 aperture layout) -- exactly what async VDST wants.
__device__ __forceinline__ unsigned lds_off(const void* p) {
  return (unsigned)(unsigned long long)p;
}

// GLOBAL_LOAD_ASYNC_TO_LDS_B128: LDS[vdst] = MEM[vaddr] (16 bytes/lane,
// tracked by ASYNCcnt).  CDNA5 ISA.
__device__ __forceinline__ void async_b128(const void* gaddr, unsigned ldsaddr) {
  asm volatile("global_load_async_to_lds_b128 %0, %1, off"
               :: "v"(ldsaddr), "v"(gaddr) : "memory");
}
__device__ __forceinline__ void wait_async0() {
  asm volatile("s_wait_asynccnt 0x0" ::: "memory");
}

// A fragment, 16x32 bf16 (contiguous k order, matches load_b32): lane = 16*khalf + row.
// elements 0..15 -> K = kb..kb+15 (kb = 16*khalf)
__device__ __forceinline__ v16bf load_a32(const __bf16* base, int lda, int lane) {
  const int row = lane & 15;
  const int kb  = (lane >> 4) << 4;
  const __bf16* p = base + row * lda + kb;
  v8bf lo = *reinterpret_cast<const v8bf*>(p);
  v8bf hi = *reinterpret_cast<const v8bf*>(p + 8);
  v16bf f;
#pragma unroll
  for (int i = 0; i < 8; ++i) { f[i] = lo[i]; f[i + 8] = hi[i]; }
  return f;
}

// B fragment, 32x16 bf16: lane = 16*khalf + col, 16 contiguous K per lane when
// B is stored N-major (B[k][n] at base[n*ldb + k]).
__device__ __forceinline__ v16bf load_b32(const __bf16* base, int ldb, int lane) {
  const int col = lane & 15;
  const int kb  = (lane >> 4) << 4;
  const __bf16* p = base + col * ldb + kb;
  v8bf lo = *reinterpret_cast<const v8bf*>(p);
  v8bf hi = *reinterpret_cast<const v8bf*>(p + 8);
  v16bf f;
#pragma unroll
  for (int i = 0; i < 8; ++i) { f[i] = lo[i]; f[i + 8] = hi[i]; }
  return f;
}

// ---------------------------------------------------------------------------
// f32 -> bf16 conversion (weights)
// ---------------------------------------------------------------------------
__global__ __launch_bounds__(256)
void k_cvt_bf16(const float* __restrict__ s, __bf16* __restrict__ d, int n) {
  int i = blockIdx.x * 256 + threadIdx.x;
  if (i < n) d[i] = (__bf16)s[i];
}

// ---------------------------------------------------------------------------
// LayerNorm over last dim (256), one wave32 per row, bf16 output.
// ---------------------------------------------------------------------------
__global__ __launch_bounds__(256)
void k_layernorm(const float* __restrict__ x, const float* __restrict__ g,
                 const float* __restrict__ beta, __bf16* __restrict__ out, int rows) {
  const int row  = blockIdx.x * 8 + (threadIdx.x >> 5);
  if (row >= rows) return;
  const int lane = threadIdx.x & 31;
  const float* rp = x + (size_t)row * DIMD + lane * 8;
  float v[8];
  float s = 0.f;
#pragma unroll
  for (int i = 0; i < 8; ++i) { v[i] = rp[i]; s += v[i]; }
#pragma unroll
  for (int off = 16; off >= 1; off >>= 1) s += __shfl_xor(s, off, 32);
  const float mu = s * (1.0f / DIMD);
  float q = 0.f;
#pragma unroll
  for (int i = 0; i < 8; ++i) { float d = v[i] - mu; q += d * d; }
#pragma unroll
  for (int off = 16; off >= 1; off >>= 1) q += __shfl_xor(q, off, 32);
  const float rstd = rsqrtf(q * (1.0f / DIMD) + 1e-5f);
  const int c0 = lane * 8;
  v8bf o;
#pragma unroll
  for (int i = 0; i < 8; ++i)
    o[i] = (__bf16)((v[i] - mu) * rstd * g[c0 + i] + beta[c0 + i]);
  *reinterpret_cast<v8bf*>(out + (size_t)row * DIMD + c0) = o;
}

// ---------------------------------------------------------------------------
// Generic GEMM: C[M,256] = A_bf16[M,256] @ W_bf16[n][k]^T + bias, fused epilogue.
// Block: 128 threads (4 waves). Block tile 64(M) x 256(N); wave w owns 64 cols.
// The shared 64x256 A tile (32 KB) is staged once via async global->LDS copies
// (dedups the 4x per-wave A reads), then read as ds_load_b128 fragments.
// EPI: 0 plain bf16 | 1 Vt-transposed bf16 | 2 +res -> f32 | 3 GELU bf16 | 4 +res -> f32 out
// ---------------------------------------------------------------------------
template <int EPI>
__global__ __launch_bounds__(128)
void k_gemm(const __bf16* __restrict__ A, const __bf16* __restrict__ W,
            const float* __restrict__ bias, const float* __restrict__ res,
            void* __restrict__ outp) {
  extern __shared__ char smem[];
  __bf16* sA = (__bf16*)smem;                 // [64][256] bf16 = 32 KB

  const int lane = threadIdx.x & 31;
  const int wv   = threadIdx.x >> 5;
  const int m0   = blockIdx.x * 64;
  const int n0   = wv * 64;

  // ---- cooperative async stage of the contiguous 32 KB A tile ----
  {
    const __bf16* gA = A + (size_t)m0 * DIMD;
    const unsigned lbase = lds_off(sA);
#pragma unroll
    for (int i = 0; i < 16; ++i) {
      const int g = threadIdx.x + 128 * i;    // 16-byte granule index
      async_b128(gA + g * 8, lbase + (unsigned)g * 16);
    }
    wait_async0();
    __syncthreads();
  }

  v8f z = {0.f, 0.f, 0.f, 0.f, 0.f, 0.f, 0.f, 0.f};
  v8f acc[4][4];
#pragma unroll
  for (int mi = 0; mi < 4; ++mi)
#pragma unroll
    for (int ni = 0; ni < 4; ++ni) acc[mi][ni] = z;

#pragma unroll
  for (int kt = 0; kt < 8; ++kt) {
    v16bf a[4], bw[4];
#pragma unroll
    for (int mi = 0; mi < 4; ++mi)
      a[mi] = load_a32(sA + (size_t)(16 * mi) * DIMD + kt * 32, DIMD, lane);
#pragma unroll
    for (int ni = 0; ni < 4; ++ni)
      bw[ni] = load_b32(W + (size_t)(n0 + 16 * ni) * DIMD + kt * 32, DIMD, lane);
#pragma unroll
    for (int mi = 0; mi < 4; ++mi)
#pragma unroll
      for (int ni = 0; ni < 4; ++ni)
        acc[mi][ni] = wmma_bf16(a[mi], bw[ni], acc[mi][ni]);
  }

  const int rlo = (lane >> 4) << 3;
  const int cl  = lane & 15;
#pragma unroll
  for (int mi = 0; mi < 4; ++mi) {
#pragma unroll
    for (int ni = 0; ni < 4; ++ni) {
#pragma unroll
      for (int r = 0; r < 8; ++r) {
        const int row = m0 + 16 * mi + rlo + r;
        const int col = n0 + 16 * ni + cl;
        float v = acc[mi][ni][r] + bias[col];
        if (EPI == 0) {                       // plain bf16 [M,256]
          ((__bf16*)outp)[(size_t)row * DIMD + col] = (__bf16)v;
        } else if (EPI == 1) {                // Vt[b][ch][kv] bf16
          const int b  = row >> 8;
          const int kv = row & 255;
          ((__bf16*)outp)[((size_t)(b * DIMD + col)) * NKV + kv] = (__bf16)v;
        } else if (EPI == 2) {                // + residual -> f32
          ((float*)outp)[(size_t)row * DIMD + col] =
              v + res[(size_t)row * DIMD + col];
        } else if (EPI == 3) {                // exact GELU -> bf16
          float gv = 0.5f * v * (1.0f + erff(v * 0.70710678118654752f));
          ((__bf16*)outp)[(size_t)row * DIMD + col] = (__bf16)gv;
        } else {                              // + residual -> f32 final out
          ((float*)outp)[(size_t)row * DIMD + col] =
              v + res[(size_t)row * DIMD + col];
        }
      }
    }
  }
}

// ---------------------------------------------------------------------------
// Fused attention: block = (batch, 64 q-rows), 4 waves x 16 rows, loops heads.
// Per head, the K-slice [256][64] and V-slice [64][256] (32 KB each) are async
// staged into LDS (dedups 4x per-wave reads), scores = (Q K^T)/8 + cw ->
// softmax -> bf16 probs (LDS) -> attn @ V -> ctx.  Head-mean of attn written
// to attn_out (block exclusively owns its rows).
// Dynamic LDS: scores 67584 + probs 33792 + K 32768 + V 32768 = 166912 B.
// ---------------------------------------------------------------------------
#define SROW 264
#define SMEM_SCORES (4 * 16 * SROW * 4)
#define SMEM_PROBS  (4 * 16 * SROW * 2)
#define SMEM_K      (NKV * HD * 2)
#define SMEM_V      (HD * NKV * 2)
#define SMEM_TOTAL  (SMEM_SCORES + SMEM_PROBS + SMEM_K + SMEM_V)

__global__ __launch_bounds__(128)
void k_attn(const __bf16* __restrict__ Qp, const __bf16* __restrict__ Kp,
            const __bf16* __restrict__ Vt, const float* __restrict__ cw,
            __bf16* __restrict__ ctx, float* __restrict__ attn_out) {
  extern __shared__ char smem[];
  const int lane = threadIdx.x & 31;
  const int wv   = threadIdx.x >> 5;
  const int b    = blockIdx.x >> 6;           // NQ/64 = 64 tiles per batch
  const int qt   = blockIdx.x & 63;
  const int q0   = qt * 64 + wv * 16;         // this wave's 16 q rows

  float*  sS = (float*)smem + (size_t)wv * 16 * SROW;
  __bf16* sP = (__bf16*)(smem + SMEM_SCORES) + (size_t)wv * 16 * SROW;
  __bf16* sK = (__bf16*)(smem + SMEM_SCORES + SMEM_PROBS);        // [256][64]
  __bf16* sV = (__bf16*)(smem + SMEM_SCORES + SMEM_PROBS + SMEM_K); // [64][256]

  const int rlo = (lane >> 4) << 3;
  const int cl  = lane & 15;

  for (int h = 0; h < NH; ++h) {
    // ---- async stage of this head's K and V slices into LDS ----
    {
      const __bf16* gK = Kp + (size_t)(b * NKV) * DIMD + h * HD; // strided rows
      const __bf16* gV = Vt + ((size_t)(b * DIMD + h * HD)) * NKV; // contiguous
      const unsigned kbase = lds_off(sK);
      const unsigned vbase = lds_off(sV);
#pragma unroll
      for (int i = 0; i < 16; ++i) {
        const int g   = threadIdx.x + 128 * i; // 16-byte granule index
        const int row = g >> 3;                // K row (kv index)
        const int c8  = (g & 7) * 8;           // channel offset within 64
        async_b128(gK + (size_t)row * DIMD + c8,
                   kbase + (unsigned)(row * HD + c8) * 2);
        async_b128(gV + g * 8, vbase + (unsigned)g * 16);
      }
      wait_async0();
      __syncthreads();
    }

    // ---- scores = Q K^T / 8 + cw ----
    v16bf aq[2];
#pragma unroll
    for (int kt = 0; kt < 2; ++kt)
      aq[kt] = load_a32(Qp + ((size_t)(b * NQ + q0)) * DIMD + h * HD + kt * 32,
                        DIMD, lane);
    v8f z = {0.f, 0.f, 0.f, 0.f, 0.f, 0.f, 0.f, 0.f};
#pragma unroll
    for (int nt = 0; nt < 16; ++nt) {
      v8f acc = z;
#pragma unroll
      for (int kt = 0; kt < 2; ++kt) {
        v16bf bk = load_b32(sK + (size_t)(nt * 16) * HD + kt * 32, HD, lane);
        acc = wmma_bf16(aq[kt], bk, acc);
      }
#pragma unroll
      for (int r = 0; r < 8; ++r) {
        const int lr = rlo + r;
        const int kv = nt * 16 + cl;
        float s = acc[r] * 0.125f +
                  cw[((size_t)(b * NQ + q0 + lr)) * NKV + kv];
        sS[lr * SROW + kv] = s;
      }
    }
    __syncthreads();

    // ---- softmax over Nkv=256 (lane owns 8 contiguous cols) ----
#pragma unroll 1
    for (int ri = 0; ri < 16; ++ri) {
      float v[8];
      float mx = -3.4e38f;
#pragma unroll
      for (int i = 0; i < 8; ++i) {
        v[i] = sS[ri * SROW + lane * 8 + i];
        mx = fmaxf(mx, v[i]);
      }
#pragma unroll
      for (int off = 16; off >= 1; off >>= 1) mx = fmaxf(mx, __shfl_xor(mx, off, 32));
      float se = 0.f;
#pragma unroll
      for (int i = 0; i < 8; ++i) { v[i] = __expf(v[i] - mx); se += v[i]; }
#pragma unroll
      for (int off = 16; off >= 1; off >>= 1) se += __shfl_xor(se, off, 32);
      const float inv = 1.0f / se;
      const size_t obase = ((size_t)(b * NQ + q0 + ri)) * NKV + lane * 8;
#pragma unroll
      for (int i = 0; i < 8; ++i) {
        float p = v[i] * inv;
        sP[ri * SROW + lane * 8 + i] = (__bf16)p;
        float pm = p * 0.25f;                  // mean over 4 heads
        if (h == 0) attn_out[obase + i] = pm;
        else        attn_out[obase + i] += pm;
      }
    }
    __syncthreads();

    // ---- ctx = attn @ V  (A from LDS probs, B from LDS V slice) ----
    v8f c2[4];
#pragma unroll
    for (int ni = 0; ni < 4; ++ni) c2[ni] = z;
#pragma unroll
    for (int kt = 0; kt < 8; ++kt) {
      v16bf ap = load_a32(sP + kt * 32, SROW, lane);
#pragma unroll
      for (int ni = 0; ni < 4; ++ni) {
        v16bf bv = load_b32(sV + (size_t)(ni * 16) * NKV + kt * 32, NKV, lane);
        c2[ni] = wmma_bf16(ap, bv, c2[ni]);
      }
    }
#pragma unroll
    for (int ni = 0; ni < 4; ++ni)
#pragma unroll
      for (int r = 0; r < 8; ++r)
        ctx[((size_t)(b * NQ + q0 + rlo + r)) * DIMD + h * HD + ni * 16 + cl] =
            (__bf16)c2[ni][r];
    __syncthreads();   // protect sK/sV/sS before next head restage
  }
}

// ---------------------------------------------------------------------------
// Host-side orchestration
// ---------------------------------------------------------------------------
extern "C" void kernel_launch(void* const* d_in, const int* in_sizes, int n_in,
                              void* d_out, int out_size, void* d_ws, size_t ws_size,
                              hipStream_t stream) {
  const float* node = (const float*)d_in[0];
  const float* clus = (const float*)d_in[1];
  const float* cw   = (const float*)d_in[2];
  const float* lnqg = (const float*)d_in[3];
  const float* lnqb = (const float*)d_in[4];
  const float* lnkg = (const float*)d_in[5];
  const float* lnkb = (const float*)d_in[6];
  const float* lnog = (const float*)d_in[7];
  const float* lnob = (const float*)d_in[8];
  const float* Wq = (const float*)d_in[9];  const float* bq = (const float*)d_in[10];
  const float* Wk = (const float*)d_in[11]; const float* bk = (const float*)d_in[12];
  const float* Wv = (const float*)d_in[13]; const float* bv = (const float*)d_in[14];
  const float* Wo = (const float*)d_in[15]; const float* bo = (const float*)d_in[16];
  const float* W1 = (const float*)d_in[17]; const float* b1 = (const float*)d_in[18];
  const float* W2 = (const float*)d_in[19]; const float* b2 = (const float*)d_in[20];

  const int MQ  = BATCH * NQ;    // 32768
  const int MKV = BATCH * NKV;   // 2048
  const size_t NW = (size_t)DIMD * DIMD;  // 65536 weight elems

  // workspace carve-out (~116 MB)
  char* ws = (char*)d_ws;
  size_t off = 0;
  auto carve = [&](size_t bytes) -> char* {
    char* p = ws + off;
    off += (bytes + 255) & ~(size_t)255;
    return p;
  };
  __bf16* q_ln  = (__bf16*)carve((size_t)MQ * DIMD * 2);
  __bf16* kv_ln = (__bf16*)carve((size_t)MKV * DIMD * 2);
  __bf16* Wqb   = (__bf16*)carve(NW * 2);
  __bf16* Wkb   = (__bf16*)carve(NW * 2);
  __bf16* Wvb   = (__bf16*)carve(NW * 2);
  __bf16* Wob   = (__bf16*)carve(NW * 2);
  __bf16* W1b   = (__bf16*)carve(NW * 2);
  __bf16* W2b   = (__bf16*)carve(NW * 2);
  __bf16* Qp    = (__bf16*)carve((size_t)MQ * DIMD * 2);
  __bf16* Kp    = (__bf16*)carve((size_t)MKV * DIMD * 2);
  __bf16* Vt    = (__bf16*)carve((size_t)MKV * DIMD * 2);  // [B][256ch][256kv]
  __bf16* ctx   = (__bf16*)carve((size_t)MQ * DIMD * 2);
  float*  xres  = (float*) carve((size_t)MQ * DIMD * 4);
  __bf16* xln   = (__bf16*)carve((size_t)MQ * DIMD * 2);
  __bf16* h1    = (__bf16*)carve((size_t)MQ * DIMD * 2);
  (void)ws_size; (void)in_sizes; (void)n_in; (void)out_size;

  float* out_f32  = (float*)d_out;
  float* attn_out = out_f32 + (size_t)MQ * DIMD;  // [B,Nq,Nkv]

  // 1) weights -> bf16 (layout [n][k] kept: B-fragments contiguous in k)
  k_cvt_bf16<<<(int)(NW / 256), 256, 0, stream>>>(Wq, Wqb, (int)NW);
  k_cvt_bf16<<<(int)(NW / 256), 256, 0, stream>>>(Wk, Wkb, (int)NW);
  k_cvt_bf16<<<(int)(NW / 256), 256, 0, stream>>>(Wv, Wvb, (int)NW);
  k_cvt_bf16<<<(int)(NW / 256), 256, 0, stream>>>(Wo, Wob, (int)NW);
  k_cvt_bf16<<<(int)(NW / 256), 256, 0, stream>>>(W1, W1b, (int)NW);
  k_cvt_bf16<<<(int)(NW / 256), 256, 0, stream>>>(W2, W2b, (int)NW);

  // 2) LayerNorms -> bf16 activations
  k_layernorm<<<MQ / 8, 256, 0, stream>>>(node, lnqg, lnqb, q_ln, MQ);
  k_layernorm<<<MKV / 8, 256, 0, stream>>>(clus, lnkg, lnkb, kv_ln, MKV);

  // 3) projections (WMMA, async-staged A tile: 32 KB dynamic LDS)
  k_gemm<0><<<MQ / 64, 128, 32768, stream>>>(q_ln, Wqb, bq, nullptr, Qp);
  k_gemm<0><<<MKV / 64, 128, 32768, stream>>>(kv_ln, Wkb, bk, nullptr, Kp);
  k_gemm<1><<<MKV / 64, 128, 32768, stream>>>(kv_ln, Wvb, bv, nullptr, Vt);

  // 4) fused attention (async K/V staging + WMMA + softmax + head-mean)
  k_attn<<<BATCH * (NQ / 64), 128, SMEM_TOTAL, stream>>>(Qp, Kp, Vt, cw, ctx,
                                                         attn_out);

  // 5) O-projection + residual -> x (f32)
  k_gemm<2><<<MQ / 64, 128, 32768, stream>>>(ctx, Wob, bo, node, xres);

  // 6) LN(x) -> bf16
  k_layernorm<<<MQ / 8, 256, 0, stream>>>(xres, lnog, lnob, xln, MQ);

  // 7) MLP: fc1+GELU -> h1 ; fc2 + x -> out
  k_gemm<3><<<MQ / 64, 128, 32768, stream>>>(xln, W1b, b1, nullptr, h1);
  k_gemm<4><<<MQ / 64, 128, 32768, stream>>>(h1, W2b, b2, xres, out_f32);
}
